// PCT_86019605004597
// MI455X (gfx1250) — matrix operand order repaired and checked by reading, not compiled, not run on hardware
//
// PCT (Point Cloud Transformer) forward for MI455X / gfx1250 (CDNA5, wave32).
//
// All heavy matmuls (mini-pointnet convs, transformer convs, attention Gram and
// apply, c3) run through one stride-generic batched WMMA GEMM using
// v_wmma_f32_16x16x32_f16 (f16 operands, f32 accumulate). FLOP budget is
// ~140 GFLOP of GEMM vs ~1GB of activation traffic -> f16 WMMA is the path
// the chip rewards; f32 accumulate preserves reduction accuracy.
//
// GEMM v2: LDS tiles are stored in *fragment-ready* layout (one aligned v16h
// per (tile,lane)), so operand builds are single 32-byte LDS vector loads
// instead of 16 scalar ds_load_u16 + waits. All staging bounds guards removed
// (every call site has M in {64,128,256,1024}, NC % 64 == 0, K % 32 == 0),
// eliminating exec-mask-predicated scalar global loads. Block tile 64x64 with
// 2 WMMAs per wave per K-step (shared A fragment).
//
// NOTE (compile-only environment): the B-matrix fragment packing uses the
// same k order as A; A follows the ISA 16-bit A 16x32
// table. The packing permutation lives only in the staging index math.
//
// Workspace: bump-allocated from d_ws, peak ~330 MB (two 128MB regions for the
// grouped tensors, reused across stages).

#include <hip/hip_runtime.h>
#include <hip/hip_bf16.h>

typedef __attribute__((ext_vector_type(16))) _Float16 v16h;
typedef __attribute__((ext_vector_type(8)))  float    v8f;
typedef _Float16 h16;

#define EPSV 1e-5f

// ---------------------------------------------------------------------------
// Generic batched WMMA GEMM:  C[b] = epilogue( W[b] @ X[b] )
//   W element (m,k):  W[b*wsb + m*wsm + k*wsk]       (any type TW)
//   X element (k,c):  X[b*xsb + k*xsk + c*xsc]       (any type TX)
//   C element (m,c):  C[b*csb + m*csm + c]           (any type TC)
// epilogue: +bias[m]; BN (g,b,m,v folded); optional ReLU.
// Block: 256 threads = 8 waves; block tile 64(M) x 64(N); K step 32.
// REQUIRES: M covered by grid.y*64 tiles with M % 16 == 0 and either M >= 64
// aligned or M == 64-tile-resident (all call sites: M in {64,128,256,1024}),
// NC % 64 == 0, K % 32 == 0.
// ---------------------------------------------------------------------------
template <typename TW, typename TX, typename TC>
__global__ __launch_bounds__(256) void k_gemm(
    const TW* __restrict__ W, long wsm, long wsk, long wsb,
    const TX* __restrict__ X, long xsk, long xsc, long xsb,
    TC* __restrict__ C, long csm, long csb,
    int M, int NC, int K,
    const float* __restrict__ bias,
    const float* __restrict__ bng, const float* __restrict__ bnb,
    const float* __restrict__ bnm, const float* __restrict__ bnv,
    int do_relu)
{
  // Fragment-ready LDS tiles: AsF[mi][lane], BsF[nt][lane]; one v16h each.
  __shared__ v16h AsF[4 * 32];   // 4 KB : A 64x32 as 4 16x32 fragments
  __shared__ v16h BsF[4 * 32];   // 4 KB : B 32x64 as 4 32x16 fragments

  const int tid  = threadIdx.x;
  const int bz   = blockIdx.z;
  const int m0   = blockIdx.y * 64;
  const int n0   = blockIdx.x * 64;
  const int wave = tid >> 5, lane = tid & 31;
  const int mi   = wave >> 1, nj = wave & 1;   // wave grid 4(M) x 2(N), 2 N-tiles/wave
  const int half = lane >> 4, lr = lane & 15;

  // staging decomposition (8 elements per thread per operand per K-step):
  //   tid[3:0] = lr (m or n within 16-tile)  -> consecutive cols for X: coalesced
  //   tid[4]   = half, tid[6:5] = tile id, tid[7] = jh; loop t supplies j[2:0]
  const int s_lr   = tid & 15;
  const int s_half = (tid >> 4) & 1;
  const int s_tile = (tid >> 5) & 3;
  const int s_jh   = (tid >> 7) & 1;
  const int s_lane = s_half * 16 + s_lr;

  h16* As = (h16*)AsF;
  h16* Bs = (h16*)BsF;

  v8f acc0 = {0.f, 0.f, 0.f, 0.f, 0.f, 0.f, 0.f, 0.f};
  v8f acc1 = {0.f, 0.f, 0.f, 0.f, 0.f, 0.f, 0.f, 0.f};

  const int a_gm = m0 + s_tile * 16 + s_lr;             // A row
  const int b_gc = n0 + s_tile * 16 + s_lr;             // B column

  for (int k0 = 0; k0 < K; k0 += 32) {
#pragma unroll
    for (int t = 0; t < 8; ++t) {
      const int j = s_jh * 8 + t;
      // A fragment element: lane holds row m, k per ISA 16-bit A 16x32 layout
      //   j<8 -> k = 8*half + j ; j>=8 -> k = 16 + 8*half + (j-8)
      const int a_gk = k0 + 16 * s_jh + 8 * s_half + t;
      As[((s_tile * 32 + s_lane) << 4) | j] =
          (h16)(float)W[(long)bz * wsb + (long)a_gm * wsm + (long)a_gk * wsk];
      // B fragment element: lane holds column n, k per same table order as A
      const int b_gk = k0 + 16 * s_jh + 8 * s_half + t;
      Bs[((s_tile * 32 + s_lane) << 4) | j] =
          (h16)(float)X[(long)bz * xsb + (long)b_gk * xsk + (long)b_gc * xsc];
    }
    // prefetch next X tile (emits global_prefetch_b8)
    if (k0 + 32 < K)
      __builtin_prefetch(
          (const void*)&X[(long)bz * xsb + (long)(k0 + 32) * xsk + (long)n0 * xsc], 0, 1);
    __syncthreads();

    const v16h af  = AsF[mi * 32 + lane];            // one 32B LDS vector load
    const v16h bf0 = BsF[(nj * 2 + 0) * 32 + lane];
    const v16h bf1 = BsF[(nj * 2 + 1) * 32 + lane];

    acc0 = __builtin_amdgcn_wmma_f32_16x16x32_f16(
        false, af, false, bf0, (short)0, acc0, false, false);
    acc1 = __builtin_amdgcn_wmma_f32_16x16x32_f16(
        false, af, false, bf1, (short)0, acc1, false, false);
    __syncthreads();
  }

  // C/D layout: VGPR r -> M = 8*half + r, N = lane&15
  const int gm0 = m0 + mi * 16 + half * 8;
#pragma unroll
  for (int r = 0; r < 8; ++r) {
    const int gm = gm0 + r;
    if (gm >= M) break;
    float s = 1.f, o = 0.f;
    if (bng) {
      float inv = bng[gm] * rsqrtf(bnv[gm] + EPSV);
      s = inv;
      o = bnb[gm] - bnm[gm] * inv;
    }
    if (bias) o += (bng ? bias[gm] * s : bias[gm]);
    {
      float y = acc0[r] * s + o;
      if (do_relu) y = fmaxf(y, 0.f);
      C[(long)bz * csb + (long)gm * csm + (n0 + (nj * 2 + 0) * 16 + lr)] = (TC)y;
    }
    {
      float y = acc1[r] * s + o;
      if (do_relu) y = fmaxf(y, 0.f);
      C[(long)bz * csb + (long)gm * csm + (n0 + (nj * 2 + 1) * 16 + lr)] = (TC)y;
    }
  }
}

template <typename TW, typename TX, typename TC>
static void gemm(hipStream_t st,
                 const TW* W, long wsm, long wsk, long wsb,
                 const TX* X, long xsk, long xsc, long xsb,
                 TC* C, long csm, long csb,
                 int M, int NC, int K, int batches,
                 const float* bias,
                 const float* g, const float* b, const float* m, const float* v,
                 int relu)
{
  dim3 grid((NC + 63) / 64, (M + 63) / 64, batches);
  k_gemm<TW, TX, TC><<<grid, dim3(256), 0, st>>>(
      W, wsm, wsk, wsb, X, xsk, xsc, xsb, C, csm, csb,
      M, NC, K, bias, g, b, m, v, relu);
}

// ---------------------------------------------------------------------------
// Small kernels
// ---------------------------------------------------------------------------

// initial 3->64 conv + bn + relu; out feature-major (64, B*N) f16
__global__ __launch_bounds__(256) void k_conv3(
    const float* __restrict__ x, const float* __restrict__ w,
    const float* __restrict__ g, const float* __restrict__ bb,
    const float* __restrict__ m, const float* __restrict__ v,
    h16* __restrict__ out, int B, int N)
{
  int col = blockIdx.x * 256 + threadIdx.x;
  long BN = (long)B * N;
  if (col >= BN) return;
  int o = blockIdx.y;
  int b = col / N, n = col % N;
  float a = 0.f;
#pragma unroll
  for (int c = 0; c < 3; ++c) a += w[o * 3 + c] * x[((long)b * 3 + c) * N + n];
  float inv = g[o] * rsqrtf(v[o] + EPSV);
  a = (a - m[o]) * inv + bb[o];
  out[(long)o * BN + col] = (h16)fmaxf(a, 0.f);
}

// x (B,3,N) -> xyz (B,N,3) f32
__global__ __launch_bounds__(256) void k_xyz0(
    const float* __restrict__ x, float* __restrict__ xyz, int B, int N)
{
  long i = (long)blockIdx.x * 256 + threadIdx.x;
  if (i >= (long)B * N) return;
  int b = (int)(i / N), n = (int)(i % N);
#pragma unroll
  for (int c = 0; c < 3; ++c) xyz[i * 3 + c] = x[((long)b * 3 + c) * N + n];
}

// farthest point sampling: one block per batch, matches lax.scan semantics
__global__ __launch_bounds__(256) void k_fps(
    const float* __restrict__ xyz, int* __restrict__ fidx, int N, int npoint)
{
  __shared__ float sv[256];
  __shared__ int   si[256];
  int b = blockIdx.x, t = threadIdx.x;
  float dmin[8];
#pragma unroll
  for (int i = 0; i < 8; ++i) dmin[i] = 1e10f;
  int far = 0;
  for (int it = 0; it < npoint; ++it) {
    if (t == 0) fidx[b * npoint + it] = far;
    const float* cp = &xyz[((long)b * N + far) * 3];
    float cx = cp[0], cy = cp[1], cz = cp[2];
    float best = -1.f; int bi = 0;
#pragma unroll
    for (int tt = 0; tt < 8; ++tt) {
      int n = t + tt * 256;
      if (n < N) {
        const float* p = &xyz[((long)b * N + n) * 3];
        float dx = p[0] - cx, dy = p[1] - cy, dz = p[2] - cz;
        float d  = dx * dx + dy * dy + dz * dz;
        float dm = fminf(dmin[tt], d);
        dmin[tt] = dm;
        if (dm > best) { best = dm; bi = n; }
      }
    }
    sv[t] = best; si[t] = bi;
    __syncthreads();
    for (int s = 128; s; s >>= 1) {
      if (t < s) {
        if (sv[t + s] > sv[t] || (sv[t + s] == sv[t] && si[t + s] < si[t])) {
          sv[t] = sv[t + s]; si[t] = si[t + s];
        }
      }
      __syncthreads();
    }
    far = si[0];
    __syncthreads();
  }
}

// ball query: one wave per (b, center); wave32 ballot + prefix popcount keeps
// the first 32 in-radius indices in ascending order (== sort-then-take-32).
__global__ void k_query_ball(
    const float* __restrict__ xyz, const int* __restrict__ fidx,
    int* __restrict__ idx, int N, int np, float r2)
{
  __shared__ int s[32];
  int b = blockIdx.y, j = blockIdx.x, lane = threadIdx.x;
  int fi = fidx[b * np + j];
  const float* cp = &xyz[((long)b * N + fi) * 3];
  float cx = cp[0], cy = cp[1], cz = cp[2];
  int cnt = 0;
  for (int n0 = 0; n0 < N && cnt < 32; n0 += 32) {
    int n = n0 + lane;
    bool in = false;
    if (n < N) {
      const float* p = &xyz[((long)b * N + n) * 3];
      float dx = p[0] - cx, dy = p[1] - cy, dz = p[2] - cz;
      in = (dx * dx + dy * dy + dz * dz) <= r2;
    }
    unsigned bal = (unsigned)__ballot(in);           // wave32: low 32 bits
    int pos = cnt + __popc(bal & ((1u << lane) - 1u));
    if (in && pos < 32) s[pos] = n;
    cnt += __popc(bal);
  }
  __syncthreads();
  int mval = cnt < 32 ? cnt : 32;                    // center always in-radius
  idx[((long)b * np + j) * 32 + lane] = (lane < mval) ? s[lane] : s[0];
}

// gather new_xyz
__global__ __launch_bounds__(256) void k_gather_xyz(
    const float* __restrict__ xyz, const int* __restrict__ fidx,
    float* __restrict__ nxyz, int N, int np, int B)
{
  long i = (long)blockIdx.x * 256 + threadIdx.x;
  if (i >= (long)B * np) return;
  int b = (int)(i / np);
  int fi = fidx[i];
#pragma unroll
  for (int c = 0; c < 3; ++c) nxyz[i * 3 + c] = xyz[((long)b * N + fi) * 3 + c];
}

// build grouped tensor G (2*Cf rows, B*np*32 cols) = [pts[idx]-pts[fidx]; pts[fidx]]
__global__ __launch_bounds__(256) void k_group(
    const h16* __restrict__ F, const int* __restrict__ fidx,
    const int* __restrict__ idx, h16* __restrict__ G,
    int B, int Nn, int np, int Cf)
{
  long total = (long)B * np * 32;
  long col = (long)blockIdx.x * 256 + threadIdx.x;
  if (col >= total) return;
  int s = (int)(col & 31);
  long bj = col >> 5;
  int j = (int)(bj % np), b = (int)(bj / np);
  int c = blockIdx.y;
  int gi = idx[((long)b * np + j) * 32 + s];
  int fi = fidx[b * np + j];
  long BNn = (long)B * Nn;
  float pv = (float)F[(long)c * BNn + (long)b * Nn + gi];
  float nv = (float)F[(long)c * BNn + (long)b * Nn + fi];
  G[(long)c * total + col]              = (h16)(pv - nv);
  G[((long)Cf + c) * total + col]       = (h16)nv;
}

// max over 32 samples: (C, cols*32) -> (C, cols)
__global__ __launch_bounds__(256) void k_maxpool(
    const h16* __restrict__ T, h16* __restrict__ F, long cols, int C)
{
  long col = (long)blockIdx.x * 256 + threadIdx.x;
  if (col >= cols) return;
  int c = blockIdx.y;
  const h16* p = &T[(long)c * cols * 32 + col * 32];
  float mx = -1e30f;
#pragma unroll
  for (int s = 0; s < 32; ++s) mx = fmaxf(mx, (float)p[s]);
  F[(long)c * cols + col] = (h16)mx;
}

// row softmax over 256 (one block per row)
__global__ __launch_bounds__(256) void k_softmax_row(float* __restrict__ e)
{
  __shared__ float red[256];
  long row = blockIdx.x;
  int t = threadIdx.x;
  float v = e[row * 256 + t];
  red[t] = v; __syncthreads();
  for (int s = 128; s; s >>= 1) { if (t < s) red[t] = fmaxf(red[t], red[t + s]); __syncthreads(); }
  float mx = red[0]; __syncthreads();
  float ex = __expf(v - mx);
  red[t] = ex; __syncthreads();
  for (int s = 128; s; s >>= 1) { if (t < s) red[t] += red[t + s]; __syncthreads(); }
  e[row * 256 + t] = ex / red[0];
}

// a /= (1e-9 + column_sum)  (one block per batch, thread = column m)
__global__ __launch_bounds__(256) void k_colnorm(float* __restrict__ a)
{
  int b = blockIdx.x, m = threadIdx.x;
  float s = 0.f;
  for (int n = 0; n < 256; ++n) s += a[((long)b * 256 + n) * 256 + m];
  float inv = 1.f / (1e-9f + s);
  for (int n = 0; n < 256; ++n) a[((long)b * 256 + n) * 256 + m] *= inv;
}

__global__ __launch_bounds__(256) void k_sub_h(
    const h16* __restrict__ a, const h16* __restrict__ b, h16* __restrict__ o, long n)
{
  long i = (long)blockIdx.x * 256 + threadIdx.x;
  if (i < n) o[i] = (h16)((float)a[i] - (float)b[i]);
}

__global__ __launch_bounds__(256) void k_add_h(
    const h16* __restrict__ a, const h16* __restrict__ b, h16* __restrict__ o, long n)
{
  long i = (long)blockIdx.x * 256 + threadIdx.x;
  if (i < n) o[i] = (h16)((float)a[i] + (float)b[i]);
}

// global max over the 256 points: HC (1024, B*256) -> g (B, 1024)
__global__ __launch_bounds__(256) void k_rowmax(
    const h16* __restrict__ HC, float* __restrict__ g)
{
  __shared__ float red[256];
  int c = blockIdx.x, b = blockIdx.y, t = threadIdx.x;
  red[t] = (float)HC[(long)c * 8192 + b * 256 + t];
  __syncthreads();
  for (int s = 128; s; s >>= 1) { if (t < s) red[t] = fmaxf(red[t], red[t + s]); __syncthreads(); }
  if (t == 0) g[(long)b * 1024 + c] = red[0];
}

// classifier head: per-batch block (1024 -> 512 LN leaky -> 256 LN leaky -> 10)
__global__ __launch_bounds__(256) void k_cls(
    const float* __restrict__ g,
    const float* __restrict__ w1, const float* __restrict__ b1,
    const float* __restrict__ l1g, const float* __restrict__ l1b,
    const float* __restrict__ w2, const float* __restrict__ b2,
    const float* __restrict__ l2g, const float* __restrict__ l2b,
    const float* __restrict__ w3, const float* __restrict__ b3,
    float* __restrict__ out)
{
  __shared__ float y1[512], y2[256], r1[256], r2[256];
  int b = blockIdx.x, t = threadIdx.x;
  const float* gb = &g[(long)b * 1024];
  for (int o = t; o < 512; o += 256) {
    float a = b1[o];
    for (int k = 0; k < 1024; ++k) a += gb[k] * w1[(long)o * 1024 + k];
    y1[o] = a;
  }
  __syncthreads();
  float s = 0.f, q = 0.f;
  for (int o = t; o < 512; o += 256) { float v = y1[o]; s += v; q += v * v; }
  r1[t] = s; r2[t] = q; __syncthreads();
  for (int st = 128; st; st >>= 1) { if (t < st) { r1[t] += r1[t + st]; r2[t] += r2[t + st]; } __syncthreads(); }
  float mu = r1[0] / 512.f, var = r2[0] / 512.f - mu * mu;
  float inv = rsqrtf(var + EPSV);
  __syncthreads();
  for (int o = t; o < 512; o += 256) {
    float v = (y1[o] - mu) * inv * l1g[o] + l1b[o];
    y1[o] = v >= 0.f ? v : 0.2f * v;
  }
  __syncthreads();
  {
    float a = b2[t];
    for (int k = 0; k < 512; ++k) a += y1[k] * w2[(long)t * 512 + k];
    y2[t] = a;
  }
  __syncthreads();
  { float v = y2[t]; r1[t] = v; r2[t] = v * v; }
  __syncthreads();
  for (int st = 128; st; st >>= 1) { if (t < st) { r1[t] += r1[t + st]; r2[t] += r2[t + st]; } __syncthreads(); }
  mu = r1[0] / 256.f; var = r2[0] / 256.f - mu * mu; inv = rsqrtf(var + EPSV);
  __syncthreads();
  {
    float v = (y2[t] - mu) * inv * l2g[t] + l2b[t];
    y2[t] = v >= 0.f ? v : 0.2f * v;
  }
  __syncthreads();
  if (t < 10) {
    float a = b3[t];
    for (int k = 0; k < 256; ++k) a += y2[k] * w3[(long)t * 256 + k];
    out[(long)b * 10 + t] = a;
  }
}

// ---------------------------------------------------------------------------
// Host side
// ---------------------------------------------------------------------------
struct AttnP { const float *wqk, *wv, *bv, *wt, *bt, *bng, *bnb, *bnm, *bnv; };

struct Ptrs {
  const float *x;
  const float *c1, *c2, *c3;
  const float *bn1g, *bn1b, *bn1m, *bn1v;
  const float *bn2g, *bn2b, *bn2m, *bn2v;
  const float *bn3g, *bn3b, *bn3m, *bn3v;
  const float *m1c1, *m1c2, *m1Ag, *m1Ab, *m1Am, *m1Av, *m1Bg, *m1Bb, *m1Bm, *m1Bv;
  const float *m2c1, *m2c2, *m2Ag, *m2Ab, *m2Am, *m2Av, *m2Bg, *m2Bb, *m2Bm, *m2Bv;
  const float *trw1, *trw2;
  const float *tb1g, *tb1b, *tb1m, *tb1v, *tb2g, *tb2b, *tb2m, *tb2v;
  AttnP sa[4];
  const float *cw1, *cb1, *cl1g, *cl1b, *cw2, *cb2, *cl2g, *cl2b, *cw3, *cb3;
};

static void map_params(void* const* d_in, const int* in_sizes, Ptrs& P)
{
  auto F = [&](int i) { return (const float*)d_in[i]; };
  if (in_sizes[0] == 32 * 3 * 2048) {
    // dict insertion order (x first, params in literal order)
    P.x = F(0);
    P.c1 = F(1);  P.bn1g = F(2);  P.bn1b = F(3);  P.bn1m = F(4);  P.bn1v = F(5);
    P.c2 = F(6);  P.bn2g = F(7);  P.bn2b = F(8);  P.bn2m = F(9);  P.bn2v = F(10);
    P.m1c1 = F(11); P.m1Ag = F(12); P.m1Ab = F(13); P.m1Am = F(14); P.m1Av = F(15);
    P.m1c2 = F(16); P.m1Bg = F(17); P.m1Bb = F(18); P.m1Bm = F(19); P.m1Bv = F(20);
    P.m2c1 = F(21); P.m2Ag = F(22); P.m2Ab = F(23); P.m2Am = F(24); P.m2Av = F(25);
    P.m2c2 = F(26); P.m2Bg = F(27); P.m2Bb = F(28); P.m2Bm = F(29); P.m2Bv = F(30);
    P.trw1 = F(31); P.tb1g = F(32); P.tb1b = F(33); P.tb1m = F(34); P.tb1v = F(35);
    P.trw2 = F(36); P.tb2g = F(37); P.tb2b = F(38); P.tb2m = F(39); P.tb2v = F(40);
    for (int i = 0; i < 4; ++i) {
      int o = 41 + i * 9;
      P.sa[i] = AttnP{F(o + 0), F(o + 1), F(o + 2), F(o + 3), F(o + 4),
                      F(o + 5), F(o + 6), F(o + 7), F(o + 8)};
    }
    P.c3 = F(77); P.bn3g = F(78); P.bn3b = F(79); P.bn3m = F(80); P.bn3v = F(81);
    P.cw1 = F(82); P.cb1 = F(83); P.cl1g = F(84); P.cl1b = F(85);
    P.cw2 = F(86); P.cb2 = F(87); P.cl2g = F(88); P.cl2b = F(89);
    P.cw3 = F(90); P.cb3 = F(91);
  } else {
    // jax pytree sorted-key order ('params' subtree first, 'x' last)
    P.bn1b = F(0);  P.bn1g = F(1);  P.bn1m = F(2);  P.bn1v = F(3);
    P.bn2b = F(4);  P.bn2g = F(5);  P.bn2m = F(6);  P.bn2v = F(7);
    P.bn3b = F(8);  P.bn3g = F(9);  P.bn3m = F(10); P.bn3v = F(11);
    P.c1 = F(12); P.c2 = F(13); P.c3 = F(14);
    P.cb1 = F(15); P.cb2 = F(16); P.cb3 = F(17);
    P.cl1b = F(18); P.cl1g = F(19); P.cl2b = F(20); P.cl2g = F(21);
    P.cw1 = F(22); P.cw2 = F(23); P.cw3 = F(24);
    P.m1Ab = F(25); P.m1Ag = F(26); P.m1Am = F(27); P.m1Av = F(28);
    P.m1Bb = F(29); P.m1Bg = F(30); P.m1Bm = F(31); P.m1Bv = F(32);
    P.m1c1 = F(33); P.m1c2 = F(34);
    P.m2Ab = F(35); P.m2Ag = F(36); P.m2Am = F(37); P.m2Av = F(38);
    P.m2Bb = F(39); P.m2Bg = F(40); P.m2Bm = F(41); P.m2Bv = F(42);
    P.m2c1 = F(43); P.m2c2 = F(44);
    P.tb1b = F(45); P.tb1g = F(46); P.tb1m = F(47); P.tb1v = F(48);
    P.tb2b = F(49); P.tb2g = F(50); P.tb2m = F(51); P.tb2v = F(52);
    for (int i = 0; i < 4; ++i) {
      int o = 53 + i * 9;  // bn.b, bn.g, bn.m, bn.v, bt, bv, wqk, wt, wv
      P.sa[i] = AttnP{F(o + 6), F(o + 8), F(o + 5), F(o + 7), F(o + 4),
                      F(o + 1), F(o + 0), F(o + 2), F(o + 3)};
    }
    P.trw1 = F(89); P.trw2 = F(90);
    P.x = F(91);
  }
}

static char* wsp(void* base, size_t& off, size_t bytes)
{
  char* p = (char*)base + off;
  off += (bytes + 255) & ~(size_t)255;
  return p;
}

static void run_attn(hipStream_t st, const AttnP& p, const h16* Hin, h16* Hout,
                     h16* Qf, h16* Vf, float* eA, h16* Xr, h16* Dt, h16* Yt)
{
  const long NT = 8192;  // B*256 columns, feature-major
  const float* z = nullptr;
  // Q/K projection (shared weight): Qf (64, 8192)
  gemm<float, h16, h16>(st, p.wqk, 256, 1, 0, Hin, NT, 1, 0, Qf, NT, 0,
                        64, (int)NT, 256, 1, z, z, z, z, z, 0);
  // V projection + bias: Vf (256, 8192)
  gemm<float, h16, h16>(st, p.wv, 256, 1, 0, Hin, NT, 1, 0, Vf, NT, 0,
                        256, (int)NT, 256, 1, p.bv, z, z, z, z, 0);
  // e[b] = Q_b^T Q_b  (per-batch Gram, via strides)
  gemm<h16, h16, float>(st, Qf, 1, NT, 256, Qf, NT, 1, 256, eA, 256, 65536,
                        256, 256, 64, 32, z, z, z, z, z, 0);
  k_softmax_row<<<dim3(32 * 256), dim3(256), 0, st>>>(eA);
  k_colnorm<<<dim3(32), dim3(256), 0, st>>>(eA);
  // xr[b] = V_b @ a_b
  gemm<h16, float, h16>(st, Vf, NT, 1, 256, eA, 256, 1, 65536, Xr, NT, 256,
                        256, 256, 256, 32, z, z, z, z, z, 0);
  long n = 256 * NT;
  k_sub_h<<<dim3((unsigned)((n + 255) / 256)), dim3(256), 0, st>>>(Hin, Xr, Dt, n);
  // y = relu(bn(wt @ (x - xr) + bt))
  gemm<float, h16, h16>(st, p.wt, 256, 1, 0, Dt, NT, 1, 0, Yt, NT, 0,
                        256, (int)NT, 256, 1, p.bt, p.bng, p.bnb, p.bnm, p.bnv, 1);
  k_add_h<<<dim3((unsigned)((n + 255) / 256)), dim3(256), 0, st>>>(Hin, Yt, Hout, n);
}

extern "C" void kernel_launch(void* const* d_in, const int* in_sizes, int n_in,
                              void* d_out, int out_size, void* d_ws, size_t ws_size,
                              hipStream_t stream)
{
  (void)n_in; (void)out_size; (void)ws_size;
  Ptrs P; map_params(d_in, in_sizes, P);

  const int B = 32, N = 2048, NP1 = 512, NP2 = 256;
  const long BN = (long)B * N;               // 65536
  const long COLS1 = (long)B * NP1 * 32;     // 524288
  const long COLS2 = (long)B * NP2 * 32;     // 262144
  const long NT = (long)B * NP2;             // 8192 transformer columns
  const float* z = nullptr;

  size_t off = 0;
  h16*   H1    = (h16*)  wsp(d_ws, off, 64 * BN * 2);
  h16*   H2    = (h16*)  wsp(d_ws, off, 64 * BN * 2);
  float* xyz0  = (float*)wsp(d_ws, off, BN * 3 * 4);
  int*   fidx1 = (int*)  wsp(d_ws, off, (size_t)B * NP1 * 4);
  int*   idx1  = (int*)  wsp(d_ws, off, (size_t)B * NP1 * 32 * 4);
  float* xyz1  = (float*)wsp(d_ws, off, (size_t)B * NP1 * 3 * 4);
  int*   fidx2 = (int*)  wsp(d_ws, off, (size_t)B * NP2 * 4);
  int*   idx2  = (int*)  wsp(d_ws, off, (size_t)B * NP2 * 32 * 4);
  h16*   BIG0  = (h16*)  wsp(d_ws, off, (size_t)128 * COLS1 * 2);  // 128MB
  h16*   BIG1  = (h16*)  wsp(d_ws, off, (size_t)128 * COLS1 * 2);  // 128MB
  h16*   F1    = (h16*)  wsp(d_ws, off, (size_t)128 * B * NP1 * 2);
  h16*   F2    = (h16*)  wsp(d_ws, off, (size_t)256 * NT * 2);
  h16*   Ht0   = (h16*)  wsp(d_ws, off, (size_t)256 * NT * 2);
  h16*   Ht1   = (h16*)  wsp(d_ws, off, (size_t)256 * NT * 2);
  h16*   Qf    = (h16*)  wsp(d_ws, off, (size_t)64 * NT * 2);
  h16*   Vf    = (h16*)  wsp(d_ws, off, (size_t)256 * NT * 2);
  h16*   Xr    = (h16*)  wsp(d_ws, off, (size_t)256 * NT * 2);
  h16*   Dt    = (h16*)  wsp(d_ws, off, (size_t)256 * NT * 2);
  h16*   Yt    = (h16*)  wsp(d_ws, off, (size_t)256 * NT * 2);
  float* eA    = (float*)wsp(d_ws, off, (size_t)B * 256 * 256 * 4);
  h16*   CAT   = (h16*)  wsp(d_ws, off, (size_t)1024 * NT * 2);
  h16*   HC    = (h16*)  wsp(d_ws, off, (size_t)1024 * NT * 2);
  float* gv    = (float*)wsp(d_ws, off, (size_t)B * 1024 * 4);

  // --- stage 0: input convs ---
  k_conv3<<<dim3((unsigned)((BN + 255) / 256), 64), dim3(256), 0, stream>>>(
      P.x, P.c1, P.bn1g, P.bn1b, P.bn1m, P.bn1v, H1, B, N);
  gemm<float, h16, h16>(stream, P.c2, 64, 1, 0, H1, BN, 1, 0, H2, BN, 0,
                        64, (int)BN, 64, 1, z, P.bn2g, P.bn2b, P.bn2m, P.bn2v, 1);

  // --- stage 1: sample_and_group(512, 0.15, 32) + mini_pointnet(128) ---
  k_xyz0<<<dim3((unsigned)((BN + 255) / 256)), dim3(256), 0, stream>>>(P.x, xyz0, B, N);
  k_fps<<<dim3(B), dim3(256), 0, stream>>>(xyz0, fidx1, N, NP1);
  k_query_ball<<<dim3(NP1, B), dim3(32), 0, stream>>>(xyz0, fidx1, idx1, N, NP1, 0.15f * 0.15f);
  k_group<<<dim3((unsigned)(COLS1 / 256), 64), dim3(256), 0, stream>>>(
      H2, fidx1, idx1, BIG0, B, N, NP1, 64);                       // G1 = BIG0 (128 x COLS1)
  gemm<float, h16, h16>(stream, P.m1c1, 128, 1, 0, BIG0, COLS1, 1, 0, BIG1, COLS1, 0,
                        128, (int)COLS1, 128, 1, z, P.m1Ag, P.m1Ab, P.m1Am, P.m1Av, 1);
  gemm<float, h16, h16>(stream, P.m1c2, 128, 1, 0, BIG1, COLS1, 1, 0, BIG0, COLS1, 0,
                        128, (int)COLS1, 128, 1, z, P.m1Bg, P.m1Bb, P.m1Bm, P.m1Bv, 1);
  k_maxpool<<<dim3((unsigned)((B * NP1) / 256), 128), dim3(256), 0, stream>>>(
      BIG0, F1, (long)B * NP1, 128);

  // --- stage 2: sample_and_group(256, 0.2, 32) + mini_pointnet(256) ---
  k_gather_xyz<<<dim3((unsigned)((B * NP1 + 255) / 256)), dim3(256), 0, stream>>>(
      xyz0, fidx1, xyz1, N, NP1, B);
  k_fps<<<dim3(B), dim3(256), 0, stream>>>(xyz1, fidx2, NP1, NP2);
  k_query_ball<<<dim3(NP2, B), dim3(32), 0, stream>>>(xyz1, fidx2, idx2, NP1, NP2, 0.2f * 0.2f);
  k_group<<<dim3((unsigned)(COLS2 / 256), 128), dim3(256), 0, stream>>>(
      F1, fidx2, idx2, BIG1, B, NP1, NP2, 128);                    // G2 = BIG1 (256 x COLS2)
  gemm<float, h16, h16>(stream, P.m2c1, 256, 1, 0, BIG1, COLS2, 1, 0, BIG0, COLS2, 0,
                        256, (int)COLS2, 256, 1, z, P.m2Ag, P.m2Ab, P.m2Am, P.m2Av, 1);
  gemm<float, h16, h16>(stream, P.m2c2, 256, 1, 0, BIG0, COLS2, 1, 0, BIG1, COLS2, 0,
                        256, (int)COLS2, 256, 1, z, P.m2Bg, P.m2Bb, P.m2Bm, P.m2Bv, 1);
  k_maxpool<<<dim3((unsigned)(NT / 256), 256), dim3(256), 0, stream>>>(BIG1, F2, NT, 256);

  // --- transformer ---
  gemm<float, h16, h16>(stream, P.trw1, 256, 1, 0, F2, NT, 1, 0, Ht0, NT, 0,
                        256, (int)NT, 256, 1, z, P.tb1g, P.tb1b, P.tb1m, P.tb1v, 1);
  gemm<float, h16, h16>(stream, P.trw2, 256, 1, 0, Ht0, NT, 1, 0, Ht1, NT, 0,
                        256, (int)NT, 256, 1, z, P.tb2g, P.tb2b, P.tb2m, P.tb2v, 1);

  run_attn(stream, P.sa[0], Ht1,               CAT + 0 * 256 * NT, Qf, Vf, eA, Xr, Dt, Yt);
  run_attn(stream, P.sa[1], CAT + 0 * 256 * NT, CAT + 1 * 256 * NT, Qf, Vf, eA, Xr, Dt, Yt);
  run_attn(stream, P.sa[2], CAT + 1 * 256 * NT, CAT + 2 * 256 * NT, Qf, Vf, eA, Xr, Dt, Yt);
  run_attn(stream, P.sa[3], CAT + 2 * 256 * NT, CAT + 3 * 256 * NT, Qf, Vf, eA, Xr, Dt, Yt);

  // --- c3 + global max + classifier ---
  gemm<float, h16, h16>(stream, P.c3, 1024, 1, 0, CAT, NT, 1, 0, HC, NT, 0,
                        1024, (int)NT, 1024, 1, z, P.bn3g, P.bn3b, P.bn3m, P.bn3v, 1);
  k_rowmax<<<dim3(1024, B), dim3(256), 0, stream>>>(HC, gv);
  k_cls<<<dim3(B), dim3(256), 0, stream>>>(
      gv, P.cw1, P.cb1, P.cl1g, P.cl1b, P.cw2, P.cb2, P.cl2g, P.cl2b,
      P.cw3, P.cb3, (float*)d_out);
}
